// Pointnet2Backbone_42554535969436
// MI455X (gfx1250) — compile-verified, matrix operand order checked by reading, not run on hardware
//
#include <hip/hip_runtime.h>
#include <hip/hip_bf16.h>

typedef __attribute__((ext_vector_type(16))) _Float16     v16h;
typedef __attribute__((ext_vector_type(8)))  float        v8f;
typedef __attribute__((ext_vector_type(4)))  unsigned int v4u;

union HFrag { v16h v; v4u q[2]; };

// ---------------------------------------------------------------------------
// Generic WMMA GEMM layer:  dst = relu(src[rows x cinP] @ W[cinP x cout] + b)
// src: f16, row-major, stride srcStride (elements, multiple of 8 -> every
//      per-lane fragment run is 16B aligned). May be LDS or global.
// Wt : f16 global, transposed layout [cout][cinP] (so W[k][n] == Wt[n*cinP+k]).
// Output either f16 (dstH) or f32 (dstF), row-major with stride dstStride.
// rows, cout multiples of 16; cinP multiple of 32. Whole block cooperates.
//
// 16-bit A-fragment per-lane layout (CDNA5 ISA): for lane half
// h = lane/16, regs 0..3 hold K = kb + 8h + {0,2,4,6}(+1), regs 4..7 hold
// K = kb + 16 + 8h + {0,2,4,6}(+1) -> two contiguous 16-byte runs per lane,
// fetched as two b128 loads (ds_load_b128 / global_load_b128).
// ---------------------------------------------------------------------------
template<bool OUT_F32>
__device__ __forceinline__ void gemm_layer(const _Float16* __restrict__ src, int srcStride,
                                           const _Float16* __restrict__ Wt,
                                           const float* __restrict__ bias,
                                           _Float16* dstH, float* dstF, int dstStride,
                                           int rows, int cinP, int cout)
{
    const int wave   = threadIdx.x >> 5;
    const int lane   = threadIdx.x & 31;
    const int nWaves = blockDim.x >> 5;
    const int mtiles = rows >> 4, ntiles = cout >> 4, ktiles = cinP >> 5;
    const int half   = lane >> 4, l16 = lane & 15;

    for (int t = wave; t < mtiles * ntiles; t += nWaves) {
        const int mt = t / ntiles, nt = t % ntiles;
        const int m = mt * 16 + l16;
        const int n = nt * 16 + l16;
        const _Float16* aBase = src + (size_t)m * srcStride + half * 8;
        const _Float16* bBase = Wt  + (size_t)n * cinP     + half * 8;
        v8f acc = {0.f, 0.f, 0.f, 0.f, 0.f, 0.f, 0.f, 0.f};
        for (int kt = 0; kt < ktiles; ++kt) {
            const int kb = kt * 32;
            HFrag A, Bm;
            A.q[0]  = *(const v4u*)(aBase + kb);
            A.q[1]  = *(const v4u*)(aBase + kb + 16);
            Bm.q[0] = *(const v4u*)(bBase + kb);
            Bm.q[1] = *(const v4u*)(bBase + kb + 16);
            acc = __builtin_amdgcn_wmma_f32_16x16x32_f16(false, A.v, false, Bm.v,
                                                         (short)0, acc, false, false);
        }
        const float bv = bias[n];
#pragma unroll
        for (int i = 0; i < 8; ++i) {
            const int r = mt * 16 + i + half * 8;   // C/D layout: M = i + 8*(lane/16)
            float val = acc[i] + bv;
            val = val > 0.f ? val : 0.f;
            if (OUT_F32) dstF[(size_t)r * dstStride + n] = val;
            else         dstH[(size_t)r * dstStride + n] = (_Float16)val;
        }
    }
}

// ---------------------------------------------------------------------------
// Weight conversion: W f32 [cin][cout] -> Wt f16 [cout][cinP] (zero-padded K)
// ---------------------------------------------------------------------------
__global__ void convert_w_kernel(const float* __restrict__ W, _Float16* __restrict__ Wt,
                                 int cin, int cout, int cinP)
{
    int n = blockIdx.x;
    for (int k = threadIdx.x; k < cinP; k += blockDim.x)
        Wt[(size_t)n * cinP + k] = (_Float16)((k < cin) ? W[(size_t)k * cout + n] : 0.f);
}

// ---------------------------------------------------------------------------
// Farthest point sampling, fully LDS-resident (SoA xyz + dists in dynamic LDS,
// up to 16*N = 320000 B <= 320KB/WGP). One block per batch; wave32 shuffle
// argmax (tie -> lowest index, matching jnp.argmax), 2 barriers/iteration.
// ---------------------------------------------------------------------------
__global__ void __launch_bounds__(1024) fps_kernel(const float* __restrict__ xyz, int N,
                                                   int npoint, int* __restrict__ inds)
{
    extern __shared__ float smem[];
    float* sx   = smem;
    float* sy   = sx + N;
    float* sz   = sy + N;
    float* dist = sz + N;
    __shared__ float rv[32];
    __shared__ int   ri[32];
    __shared__ int   s_last;

    int b = blockIdx.x;
    const float* p = xyz + (size_t)b * N * 3;
    int tid = threadIdx.x, nt = blockDim.x;

    for (int i = tid; i < N; i += nt) {
        sx[i] = p[i * 3 + 0];
        sy[i] = p[i * 3 + 1];
        sz[i] = p[i * 3 + 2];
        dist[i] = 1e10f;
    }
    if (tid == 0) { inds[(size_t)b * npoint] = 0; s_last = 0; }
    __syncthreads();

    int lane = tid & 31, wv = tid >> 5;
    for (int it = 1; it < npoint; ++it) {
        int last = s_last;
        float lx = sx[last], ly = sy[last], lz = sz[last];
        float bestV = -1.f; int bestI = 0x7fffffff;
        for (int i = tid; i < N; i += nt) {
            float dx = sx[i] - lx, dy = sy[i] - ly, dz = sz[i] - lz;
            float d = dx * dx + dy * dy + dz * dz;
            float dd = dist[i];
            d = d < dd ? d : dd;
            dist[i] = d;
            if (d > bestV) { bestV = d; bestI = i; }   // strict > keeps first max
        }
#pragma unroll
        for (int o = 16; o > 0; o >>= 1) {
            float ov = __shfl_down(bestV, o);
            int   oi = __shfl_down(bestI, o);
            if (ov > bestV || (ov == bestV && oi < bestI)) { bestV = ov; bestI = oi; }
        }
        if (lane == 0) { rv[wv] = bestV; ri[wv] = bestI; }
        __syncthreads();
        if (tid < 32) {
            bestV = rv[tid]; bestI = ri[tid];
#pragma unroll
            for (int o = 16; o > 0; o >>= 1) {
                float ov = __shfl_down(bestV, o);
                int   oi = __shfl_down(bestI, o);
                if (ov > bestV || (ov == bestV && oi < bestI)) { bestV = ov; bestI = oi; }
            }
            if (tid == 0) { inds[(size_t)b * npoint + it] = bestI; s_last = bestI; }
        }
        __syncthreads();
    }
}

__global__ void gather_xyz_kernel(const float* __restrict__ xyz, int N,
                                  const int* __restrict__ inds, int npoint,
                                  float* __restrict__ out, int total)
{
    int t = blockIdx.x * blockDim.x + threadIdx.x;
    if (t >= total) return;
    int b = t / npoint;
    int s = inds[t];
    const float* src = xyz + ((size_t)b * N + s) * 3;
    out[(size_t)t * 3 + 0] = src[0];
    out[(size_t)t * 3 + 1] = src[1];
    out[(size_t)t * 3 + 2] = src[2];
}

// ---------------------------------------------------------------------------
// Ball query, LDS-tiled: each block = 256 queries of one batch; source cloud
// streamed through 1024-point SoA tiles in LDS. Smallest indices within
// radius, pad with first (reference semantics). Grid is exact (npoint % 256
// == 0), so barrier participation is uniform.
// ---------------------------------------------------------------------------
#define BQ_TILE 1024
__global__ void __launch_bounds__(256) ball_query_kernel(
    const float* __restrict__ xyz, int N,
    const float* __restrict__ new_xyz, int npoint,
    float r2, int nsample, int* __restrict__ idx)
{
    __shared__ float tx[BQ_TILE], ty[BQ_TILE], tz[BQ_TILE];
    int q = blockIdx.x * 256 + threadIdx.x;         // query id
    int b = q / npoint;                             // uniform per block
    float qx = new_xyz[(size_t)q * 3], qy = new_xyz[(size_t)q * 3 + 1], qz = new_xyz[(size_t)q * 3 + 2];
    const float* p = xyz + (size_t)b * N * 3;
    int* out = idx + (size_t)q * nsample;
    int cnt = 0, first = 0;

    for (int base = 0; base < N; base += BQ_TILE) {
        int tile = N - base; if (tile > BQ_TILE) tile = BQ_TILE;
        __syncthreads();
        for (int i = threadIdx.x; i < tile; i += 256) {
            tx[i] = p[(size_t)(base + i) * 3 + 0];
            ty[i] = p[(size_t)(base + i) * 3 + 1];
            tz[i] = p[(size_t)(base + i) * 3 + 2];
        }
        __syncthreads();
        if (cnt < nsample) {
            for (int i = 0; i < tile && cnt < nsample; ++i) {
                float dx = tx[i] - qx, dy = ty[i] - qy, dz = tz[i] - qz;
                float d = dx * dx + dy * dy + dz * dz;
                if (d <= r2) { if (cnt == 0) first = base + i; out[cnt++] = base + i; }
            }
        }
    }
    for (; cnt < nsample; ++cnt) out[cnt] = first;
}

// ---------------------------------------------------------------------------
// Fused SA: gather group -> 3-layer WMMA MLP in LDS -> max over samples
// One block (256 thr / 8 waves) per group.
// ---------------------------------------------------------------------------
__global__ void __launch_bounds__(256) sa_kernel(
    const float* __restrict__ xyz, int Nsrc,
    const float* __restrict__ feats, int Cfeat,
    const float* __restrict__ new_xyz, const int* __restrict__ idx,
    int npoint, int nsample, float inv_radius, int cin_raw, int cinP,
    const _Float16* __restrict__ Wt1, const float* __restrict__ b1, int c1,
    const _Float16* __restrict__ Wt2, const float* __restrict__ b2, int c2,
    const _Float16* __restrict__ Wt3, const float* __restrict__ b3, int c3,
    float* __restrict__ fout)
{
    __shared__ _Float16 X[5120];   // max(rows*cinP, rows*c2)
    __shared__ _Float16 Y[8192];   // max(rows*c1,  rows*c3)
    __shared__ int sgidx[64];
    int g = blockIdx.x;            // b*npoint + j
    int b = g / npoint;
    float cx = new_xyz[(size_t)g * 3], cy = new_xyz[(size_t)g * 3 + 1], cz = new_xyz[(size_t)g * 3 + 2];
    const int* gidx = idx + (size_t)g * nsample;
    if (threadIdx.x < nsample) sgidx[threadIdx.x] = gidx[threadIdx.x];
    __syncthreads();

    for (int t = threadIdx.x; t < nsample * cinP; t += blockDim.x) {
        int r = t / cinP, c = t % cinP;
        int s = sgidx[r];
        float val = 0.f;
        if (c < 3) {
            float pc = xyz[((size_t)b * Nsrc + s) * 3 + c];
            float ctr = (c == 0) ? cx : (c == 1) ? cy : cz;
            val = (pc - ctr) * inv_radius;
        } else if (c < cin_raw) {
            val = feats[((size_t)b * Nsrc + s) * Cfeat + (c - 3)];
        }
        X[t] = (_Float16)val;
    }
    __syncthreads();
    gemm_layer<false>(X, cinP, Wt1, b1, Y, nullptr, c1, nsample, cinP, c1);
    __syncthreads();
    gemm_layer<false>(Y, c1, Wt2, b2, X, nullptr, c2, nsample, c1, c2);
    __syncthreads();
    gemm_layer<false>(X, c2, Wt3, b3, Y, nullptr, c3, nsample, c2, c3);
    __syncthreads();
    for (int c = threadIdx.x; c < c3; c += blockDim.x) {
        float m = -3e38f;
        for (int r = 0; r < nsample; ++r) {
            float v = (float)Y[r * c3 + c];
            m = v > m ? v : m;
        }
        fout[(size_t)g * c3 + c] = m;
    }
}

// ---------------------------------------------------------------------------
// FP interpolation: 3-NN inverse-distance interp; writes f16 MLP input rows
// [interp(256) | unknown_feats(256)]. One block per unknown point.
// ---------------------------------------------------------------------------
__global__ void __launch_bounds__(256) fp_interp_kernel(
    const float* __restrict__ uxyz, int nU,
    const float* __restrict__ kxyz, int nK,
    const float* __restrict__ kfeats, const float* __restrict__ ufeats,
    _Float16* __restrict__ xin)
{
    const int C = 256;
    __shared__ float sv[64 * 3];
    __shared__ int   si[64 * 3];
    __shared__ float sw[3];
    __shared__ int   sid[3];
    int g = blockIdx.x;
    int b = g / nU;
    float qx = uxyz[(size_t)g * 3], qy = uxyz[(size_t)g * 3 + 1], qz = uxyz[(size_t)g * 3 + 2];
    const float* kp = kxyz + (size_t)b * nK * 3;
    int tid = threadIdx.x;
    if (tid < 64) {
        float v0 = -3e38f, v1 = -3e38f, v2 = -3e38f;
        int   i0 = 1 << 30, i1 = 1 << 30, i2 = 1 << 30;
        for (int i = tid; i < nK; i += 64) {
            float dx = kp[i * 3] - qx, dy = kp[i * 3 + 1] - qy, dz = kp[i * 3 + 2] - qz;
            float nd = -(dx * dx + dy * dy + dz * dz);
            if (nd > v0 || (nd == v0 && i < i0))      { v2=v1;i2=i1; v1=v0;i1=i0; v0=nd;i0=i; }
            else if (nd > v1 || (nd == v1 && i < i1)) { v2=v1;i2=i1; v1=nd;i1=i; }
            else if (nd > v2 || (nd == v2 && i < i2)) { v2=nd;i2=i; }
        }
        sv[tid*3]=v0; sv[tid*3+1]=v1; sv[tid*3+2]=v2;
        si[tid*3]=i0; si[tid*3+1]=i1; si[tid*3+2]=i2;
    }
    __syncthreads();
    if (tid == 0) {
        float v0 = -3e38f, v1 = -3e38f, v2 = -3e38f;
        int   i0 = 1 << 30, i1 = 1 << 30, i2 = 1 << 30;
        for (int c = 0; c < 64 * 3; ++c) {
            float nd = sv[c]; int i = si[c];
            if (i >= (1 << 30)) continue;
            if (nd > v0 || (nd == v0 && i < i0))      { v2=v1;i2=i1; v1=v0;i1=i0; v0=nd;i0=i; }
            else if (nd > v1 || (nd == v1 && i < i1)) { v2=v1;i2=i1; v1=nd;i1=i; }
            else if (nd > v2 || (nd == v2 && i < i2)) { v2=nd;i2=i; }
        }
        float w0 = 1.f / (-v0 + 1e-8f);
        float w1 = 1.f / (-v1 + 1e-8f);
        float w2 = 1.f / (-v2 + 1e-8f);
        float s = w0 + w1 + w2;
        sw[0] = w0 / s; sw[1] = w1 / s; sw[2] = w2 / s;
        sid[0] = i0; sid[1] = i1; sid[2] = i2;
    }
    __syncthreads();
    const float* kf = kfeats + (size_t)b * nK * C;
    const float* uf = ufeats + (size_t)g * C;
    _Float16* xr = xin + (size_t)g * (2 * C);
    for (int c = tid; c < C; c += blockDim.x) {
        float v = sw[0] * kf[(size_t)sid[0] * C + c]
                + sw[1] * kf[(size_t)sid[1] * C + c]
                + sw[2] * kf[(size_t)sid[2] * C + c];
        xr[c]     = (_Float16)v;
        xr[C + c] = (_Float16)uf[c];
    }
}

// ---------------------------------------------------------------------------
// FP MLP: 512 -> 256 -> 256 (relu both). 32 rows per block, 8 waves.
// A-fragments read straight from global f16 input; mid activation in LDS.
// ---------------------------------------------------------------------------
__global__ void __launch_bounds__(256) fp_mlp_kernel(
    const _Float16* __restrict__ X,
    const _Float16* __restrict__ Wt1, const float* __restrict__ b1,
    const _Float16* __restrict__ Wt2, const float* __restrict__ b2,
    float* __restrict__ out)
{
    __shared__ _Float16 mid[32 * 256];
    const _Float16* x = X + (size_t)blockIdx.x * 32 * 512;
    gemm_layer<false>(x, 512, Wt1, b1, mid, nullptr, 256, 32, 512, 256);
    __syncthreads();
    float* o = out + (size_t)blockIdx.x * 32 * 256;
    gemm_layer<true>(mid, 256, Wt2, b2, nullptr, o, 256, 32, 256, 256);
}

// ---------------------------------------------------------------------------
// Final output assembly
// ---------------------------------------------------------------------------
__global__ void finalize_kernel(const float* __restrict__ f2u, const float* __restrict__ xyz2,
                                const int* __restrict__ inds1, float* __restrict__ out, int total)
{
    const int Bb = 4, NP = 1024, C = 256;
    int t = blockIdx.x * blockDim.x + threadIdx.x;
    if (t >= total) return;
    const int featTot = Bb * C * NP;
    const int xyzTot  = Bb * NP * 3;
    if (t < featTot) {
        int b = t / (C * NP); int rem = t % (C * NP);
        int c = rem / NP; int n = rem % NP;
        out[t] = f2u[((size_t)b * NP + n) * C + c];        // (B,256,1024) transpose
    } else if (t < featTot + xyzTot) {
        out[t] = xyz2[t - featTot];
    } else {
        int u = t - featTot - xyzTot;
        int b = u / NP, n = u % NP;
        out[t] = (float)inds1[(size_t)b * 2048 + n];       // fp2_inds = inds1[:, :1024]
    }
}

// ---------------------------------------------------------------------------
extern "C" void kernel_launch(void* const* d_in, const int* in_sizes, int n_in,
                              void* d_out, int out_size, void* d_ws, size_t ws_size,
                              hipStream_t stream)
{
    const int Bb = 4, N = 20000;
    // Canonical layer order: sa1(0..2) sa2(3..5) sa3(6..8) sa4(9..11) fp1(12,13) fp2(14,15)
    static const int LCIN [16] = {6,64,64, 131,128,128, 259,128,128, 259,128,128, 512,256, 512,256};
    static const int LCOUT[16] = {64,64,128, 128,128,256, 128,128,256, 128,128,256, 256,256, 256,256};
    auto pad32 = [](int x) { return (x + 31) & ~31; };

    // --- detect input flattening order ---
    int pcIdx, ftIdx, pstart;
    if (n_in >= 2 && in_sizes[1] == Bb * N * 3) { pcIdx = 0; ftIdx = 1; pstart = 2; }     // insertion order
    else { ftIdx = 0; pcIdx = n_in - 1; pstart = 1; }                                     // sorted-key order
    bool saFirst = (in_sizes[pstart] == 6 * 64);
    int mapOrder[16];
    if (saFirst) { for (int i = 0; i < 16; ++i) mapOrder[i] = i; }
    else {
        static const int fpFirst[16] = {12,13,14,15, 0,1,2,3,4,5,6,7,8,9,10,11};
        for (int i = 0; i < 16; ++i) mapOrder[i] = fpFirst[i];
    }
    const float* pc    = (const float*)d_in[pcIdx];
    const float* feats = (const float*)d_in[ftIdx];
    const float* Wf[16]; const float* Bf[16];
    for (int s = 0; s < 16; ++s) {
        int c = mapOrder[s];
        Wf[c] = (const float*)d_in[pstart + 2 * s];
        Bf[c] = (const float*)d_in[pstart + 2 * s + 1];
    }

    // --- scratch bump allocator ---
    size_t off = 0; char* wsb = (char*)d_ws;
    auto alloc = [&](size_t bytes) -> void* {
        off = (off + 255) & ~(size_t)255;
        void* p = wsb + off; off += bytes; return p;
    };
    _Float16* Wt[16];
    for (int l = 0; l < 16; ++l)
        Wt[l] = (_Float16*)alloc((size_t)LCOUT[l] * pad32(LCIN[l]) * sizeof(_Float16));
    int* inds1 = (int*)alloc(sizeof(int) * Bb * 2048);
    int* inds2 = (int*)alloc(sizeof(int) * Bb * 1024);
    int* inds3 = (int*)alloc(sizeof(int) * Bb * 512);
    int* inds4 = (int*)alloc(sizeof(int) * Bb * 256);
    float* nx1 = (float*)alloc(sizeof(float) * Bb * 2048 * 3);
    float* nx2 = (float*)alloc(sizeof(float) * Bb * 1024 * 3);
    float* nx3 = (float*)alloc(sizeof(float) * Bb * 512 * 3);
    float* nx4 = (float*)alloc(sizeof(float) * Bb * 256 * 3);
    int* idx1 = (int*)alloc(sizeof(int) * Bb * 2048 * 64);
    int* idx2 = (int*)alloc(sizeof(int) * Bb * 1024 * 32);
    int* idx3 = (int*)alloc(sizeof(int) * Bb * 512 * 16);
    int* idx4 = (int*)alloc(sizeof(int) * Bb * 256 * 16);
    float* f1 = (float*)alloc(sizeof(float) * Bb * 2048 * 128);
    float* f2 = (float*)alloc(sizeof(float) * Bb * 1024 * 256);
    float* f3 = (float*)alloc(sizeof(float) * Bb * 512 * 256);
    float* f4 = (float*)alloc(sizeof(float) * Bb * 256 * 256);
    _Float16* xin1 = (_Float16*)alloc(sizeof(_Float16) * Bb * 512 * 512);
    _Float16* xin2 = (_Float16*)alloc(sizeof(_Float16) * Bb * 1024 * 512);
    float* f3u = (float*)alloc(sizeof(float) * Bb * 512 * 256);
    float* f2u = (float*)alloc(sizeof(float) * Bb * 1024 * 256);

    // --- weight conversion (f32 [cin][cout] -> f16 [cout][cinP]) ---
    for (int l = 0; l < 16; ++l)
        convert_w_kernel<<<LCOUT[l], 128, 0, stream>>>(Wf[l], Wt[l], LCIN[l], LCOUT[l], pad32(LCIN[l]));

    // --- SA1: 20000 -> 2048, r=0.04, ns=64, 6->64->64->128 ---
    fps_kernel<<<Bb, 1024, 16 * (size_t)N, stream>>>(pc, N, 2048, inds1);
    gather_xyz_kernel<<<(Bb*2048+255)/256, 256, 0, stream>>>(pc, N, inds1, 2048, nx1, Bb*2048);
    ball_query_kernel<<<(Bb*2048)/256, 256, 0, stream>>>(pc, N, nx1, 2048, 0.04f*0.04f, 64, idx1);
    sa_kernel<<<Bb*2048, 256, 0, stream>>>(pc, N, feats, 3, nx1, idx1, 2048, 64, 1.f/0.04f, 6, 32,
                                           Wt[0], Bf[0], 64, Wt[1], Bf[1], 64, Wt[2], Bf[2], 128, f1);

    // --- SA2: 2048 -> 1024, r=0.1, ns=32, 131->128->128->256 ---
    fps_kernel<<<Bb, 1024, 16 * 2048, stream>>>(nx1, 2048, 1024, inds2);
    gather_xyz_kernel<<<(Bb*1024+255)/256, 256, 0, stream>>>(nx1, 2048, inds2, 1024, nx2, Bb*1024);
    ball_query_kernel<<<(Bb*1024)/256, 256, 0, stream>>>(nx1, 2048, nx2, 1024, 0.1f*0.1f, 32, idx2);
    sa_kernel<<<Bb*1024, 256, 0, stream>>>(nx1, 2048, f1, 128, nx2, idx2, 1024, 32, 1.f/0.1f, 131, 160,
                                           Wt[3], Bf[3], 128, Wt[4], Bf[4], 128, Wt[5], Bf[5], 256, f2);

    // --- SA3: 1024 -> 512, r=0.2, ns=16, 259->128->128->256 ---
    fps_kernel<<<Bb, 1024, 16 * 1024, stream>>>(nx2, 1024, 512, inds3);
    gather_xyz_kernel<<<(Bb*512+255)/256, 256, 0, stream>>>(nx2, 1024, inds3, 512, nx3, Bb*512);
    ball_query_kernel<<<(Bb*512)/256, 256, 0, stream>>>(nx2, 1024, nx3, 512, 0.2f*0.2f, 16, idx3);
    sa_kernel<<<Bb*512, 256, 0, stream>>>(nx2, 1024, f2, 256, nx3, idx3, 512, 16, 1.f/0.2f, 259, 288,
                                          Wt[6], Bf[6], 128, Wt[7], Bf[7], 128, Wt[8], Bf[8], 256, f3);

    // --- SA4: 512 -> 256, r=0.3, ns=16, 259->128->128->256 ---
    fps_kernel<<<Bb, 1024, 16 * 512, stream>>>(nx3, 512, 256, inds4);
    gather_xyz_kernel<<<(Bb*256+255)/256, 256, 0, stream>>>(nx3, 512, inds4, 256, nx4, Bb*256);
    ball_query_kernel<<<(Bb*256)/256, 256, 0, stream>>>(nx3, 512, nx4, 256, 0.3f*0.3f, 16, idx4);
    sa_kernel<<<Bb*256, 256, 0, stream>>>(nx3, 512, f3, 256, nx4, idx4, 256, 16, 1.f/0.3f, 259, 288,
                                          Wt[9], Bf[9], 128, Wt[10], Bf[10], 128, Wt[11], Bf[11], 256, f4);

    // --- FP1: interp f4 @ xyz4 onto xyz3, concat f3, MLP 512->256->256 ---
    fp_interp_kernel<<<Bb*512, 256, 0, stream>>>(nx3, 512, nx4, 256, f4, f3, xin1);
    fp_mlp_kernel<<<(Bb*512)/32, 256, 0, stream>>>(xin1, Wt[12], Bf[12], Wt[13], Bf[13], f3u);

    // --- FP2: interp f3u @ xyz3 onto xyz2, concat f2, MLP 512->256->256 ---
    fp_interp_kernel<<<Bb*1024, 256, 0, stream>>>(nx2, 1024, nx3, 512, f3u, f2, xin2);
    fp_mlp_kernel<<<(Bb*1024)/32, 256, 0, stream>>>(xin2, Wt[14], Bf[14], Wt[15], Bf[15], f2u);

    // --- outputs: features (B,256,1024), xyz2 (B,1024,3), fp2_inds (B,1024) ---
    int total = Bb*256*1024 + Bb*1024*3 + Bb*1024;
    if (total > out_size) total = out_size;
    finalize_kernel<<<(total+255)/256, 256, 0, stream>>>(f2u, nx2, inds1, (float*)d_out, total);
}
